// Attention_7748121002186
// MI455X (gfx1250) — matrix operand order repaired and checked by reading, not compiled, not run on hardware
//
#include <hip/hip_runtime.h>

// ---------------------------------------------------------------------------
// Single-head attention for MI455X (gfx1250, wave32, WMMA).
//   q,k,v = x @ W[0..2];  s = q k^T / 8;  p = softmax(s);  out = p v
// All GEMMs: block = 8 waves -> 128x128 C tile, B tile staged in LDS
// (double-buffered, bank-conflict-free stride-72 layout), A direct from L2.
// ~64 FLOP per L2 byte per block; intermediates are L2-resident (192MB).
// ---------------------------------------------------------------------------

typedef __attribute__((ext_vector_type(16))) _Float16 v16h;
typedef __attribute__((ext_vector_type(8)))  _Float16 v8h;
typedef __attribute__((ext_vector_type(8)))  float    v8f;

namespace {
constexpr int kB = 4, kN = 2048, kD = 768, kO = 768;
constexpr size_t XH_ELEMS  = (size_t)kB * kN * kD;   // x in f16
constexpr size_t WT_ELEMS  = (size_t)3 * kO * kD;    // W transposed [m][o][d], f16
constexpr size_t QKV_ELEMS = (size_t)kB * kN * kO;   // each of q, k, vt
constexpr size_t SC_ELEMS  = (size_t)kB * kN * kN;   // scores / probs, f16
constexpr int LDS_STRIDE = 72;                       // halves; conflict-free
constexpr int LDS_BUF    = 128 * LDS_STRIDE;         // one B tile buffer (halves)
static_assert(kN % 128 == 0 && kO % 128 == 0 && kD % 32 == 0, "tiling");
}

// ---- WMMA fragment loaders (CDNA5 16x16x32 f16 layouts) --------------------
// A (16x32, MxK): lanes 0-15 row M=lane, halves {K0..K0+7, K0+16..K0+23};
//                 lanes 16-31 row M=lane-16, halves {K0+8..15, K0+24..31}.
__device__ __forceinline__ v16h load_a_frag(const _Float16* __restrict__ base,
                                            int ld, int row0, int k0) {
  const int lane = threadIdx.x & 31;
  const int r    = row0 + (lane & 15);
  const int kofs = k0 + ((lane >> 4) << 3);          // +0 or +8
  const _Float16* p = base + (size_t)r * ld + kofs;
  v8h lo = *(const v8h*)(p);                         // K = kofs .. kofs+7
  v8h hi = *(const v8h*)(p + 16);                    // K = kofs+16 .. kofs+23
  v16h a;
#pragma unroll
  for (int i = 0; i < 8; ++i) { a[i] = lo[i]; a[i + 8] = hi[i]; }
  return a;
}

// B (32x16, KxN) fragment from the LDS-staged tile.
// Column N = lane&15 (local col 16j+..), lanes 0-15 hold K={k..k+7,k+16..k+23},
// lanes 16-31 hold K={k+8..k+15,k+24..k+31}. LDS tile layout: [col][72-half stride].
__device__ __forceinline__ v16h load_b_frag_lds(const _Float16* lbase,
                                                int jcol0) {
  const int lane = threadIdx.x & 31;
  const _Float16* p = lbase + (jcol0 + (lane & 15)) * LDS_STRIDE +
                      ((lane >> 4) << 3);
  v8h lo = ((const v8h*)p)[0];
  v8h hi = ((const v8h*)p)[2];
  v16h b;
#pragma unroll
  for (int i = 0; i < 8; ++i) { b[i] = lo[i]; b[i + 8] = hi[i]; }
  return b;
}

// C/D (16x16 f32): VGPR r, lanes 0-15 -> (M=r, N=lane); lanes 16-31 -> (M=r+8).
__device__ __forceinline__ void store_c_f16(_Float16* __restrict__ base, int ld,
                                            int row0, int col0, const v8f c,
                                            float scale) {
  const int lane = threadIdx.x & 31;
  const int col  = col0 + (lane & 15);
  const int rb   = row0 + ((lane >> 4) << 3);
#pragma unroll
  for (int r = 0; r < 8; ++r)
    base[(size_t)(rb + r) * ld + col] = (_Float16)(c[r] * scale);
}

__device__ __forceinline__ void store_c_f16_T(_Float16* __restrict__ base, int ldT,
                                              int row0, int col0, const v8f c) {
  const int lane = threadIdx.x & 31;
  const int col  = col0 + (lane & 15);
  const int rb   = row0 + ((lane >> 4) << 3);
#pragma unroll
  for (int r = 0; r < 8; ++r)
    base[(size_t)col * ldT + (rb + r)] = (_Float16)c[r];
}

__device__ __forceinline__ void store_c_f32(float* __restrict__ base, int ld,
                                            int row0, int col0, const v8f c) {
  const int lane = threadIdx.x & 31;
  const int col  = col0 + (lane & 15);
  const int rb   = row0 + ((lane >> 4) << 3);
#pragma unroll
  for (int r = 0; r < 8; ++r)
    base[(size_t)(rb + r) * ld + col] = c[r];
}

// ---- Block GEMM: 256 threads = 8 waves, C tile 128(M) x 128(N) -------------
// Wave w owns rows [row0+16w, row0+16w+16). B tile (32 x 128) staged in LDS,
// double buffered, one barrier per k-step. A fragments direct from global/L2.
template <int K>
__device__ __forceinline__ void gemm_block_128x128(
    const _Float16* __restrict__ A, int lda, int row0,
    const _Float16* __restrict__ Bt, int ldb, int col0,
    _Float16* lds, v8f acc[8]) {
  const int tid   = threadIdx.x;
  const int wave  = tid >> 5;
  const int scol  = tid >> 1;            // staging col 0..127
  const int spart = (tid & 1) << 4;      // halves 0 or 16
  const int myrow0 = row0 + wave * 16;
  const _Float16* bsrc = Bt + (size_t)(col0 + scol) * ldb + spart;

  // Stage k-tile 0 into buffer 0.
  {
    v16h s = *(const v16h*)(bsrc);
    v8h* d = (v8h*)&lds[scol * LDS_STRIDE + spart];
    d[0] = ((const v8h*)&s)[0];
    d[1] = ((const v8h*)&s)[1];
  }
  __syncthreads();

  int p = 0;
  for (int k0 = 0; k0 < K; k0 += 32) {
    v16h nxt;
    const bool more = (k0 + 32 < K);     // wave-uniform
    if (more) nxt = *(const v16h*)(bsrc + k0 + 32);

    v16h a = load_a_frag(A, lda, myrow0, k0);
    const _Float16* lbase = lds + p * LDS_BUF;
#pragma unroll
    for (int j = 0; j < 8; ++j) {
      v16h b = load_b_frag_lds(lbase, 16 * j);
      acc[j] = __builtin_amdgcn_wmma_f32_16x16x32_f16(
          false, a, false, b, (short)0, acc[j], false, false);
    }
    if (more) {
      v8h* d = (v8h*)&lds[(1 - p) * LDS_BUF + scol * LDS_STRIDE + spart];
      d[0] = ((const v8h*)&nxt)[0];
      d[1] = ((const v8h*)&nxt)[1];
    }
    __syncthreads();                     // waits DScnt: publishes tile k0+32
    p ^= 1;
  }
}

// ---- Stage 1: f32 -> f16 convert for x, and W transpose ([m][d][o]->[m][o][d])
__global__ void __launch_bounds__(256)
prep_kernel(const float* __restrict__ x, const float* __restrict__ w,
            _Float16* __restrict__ xh, _Float16* __restrict__ wt) {
  const size_t base = (size_t)blockIdx.x * 1024 + (size_t)threadIdx.x * 4;
#pragma unroll
  for (int t = 0; t < 4; ++t) {
    size_t i = base + t;
    if (i < XH_ELEMS) {
      xh[i] = (_Float16)x[i];
    } else {
      size_t j = i - XH_ELEMS;                 // < WT_ELEMS
      int m = (int)(j / ((size_t)kO * kD));
      size_t r = j % ((size_t)kO * kD);
      int o = (int)(r / kD), d = (int)(r % kD);
      wt[j] = (_Float16)w[(size_t)m * kD * kO + (size_t)d * kO + o];
    }
  }
}

// ---- Stage 2: Q/K/V projections. 3*4*16*6 = 1152 blocks --------------------
__global__ void __launch_bounds__(256)
qkv_gemm_kernel(const _Float16* __restrict__ xh, const _Float16* __restrict__ wt,
                _Float16* __restrict__ q, _Float16* __restrict__ kk,
                _Float16* __restrict__ vt) {
  __shared__ _Float16 lds[2 * LDS_BUF];
  int id = blockIdx.x;
  const int nt = id % (kO / 128); id /= (kO / 128);
  const int mt = id % (kN / 128); id /= (kN / 128);
  const int b  = id % kB;         id /= kB;
  const int m  = id;                                  // 0=q 1=k 2=v
  const _Float16* A  = xh + (size_t)b * kN * kD;
  const _Float16* Bt = wt + (size_t)m * kO * kD;
  const int row0 = mt * 128, col0 = nt * 128;
  v8f acc[8] = {};
  gemm_block_128x128<kD>(A, kD, row0, Bt, kD, col0, lds, acc);
  const int myrow0 = row0 + ((threadIdx.x >> 5) << 4);
  if (m == 0) {
#pragma unroll
    for (int j = 0; j < 8; ++j)
      store_c_f16(q + (size_t)b * kN * kO, kO, myrow0, col0 + 16 * j, acc[j], 1.0f);
  } else if (m == 1) {
#pragma unroll
    for (int j = 0; j < 8; ++j)
      store_c_f16(kk + (size_t)b * kN * kO, kO, myrow0, col0 + 16 * j, acc[j], 1.0f);
  } else {  // V stored transposed [b][o][n] so PV B-fragments are contiguous
#pragma unroll
    for (int j = 0; j < 8; ++j)
      store_c_f16_T(vt + (size_t)b * kO * kN, kN, myrow0, col0 + 16 * j, acc[j]);
  }
}

// ---- Stage 3: scores = Q K^T / 8, stored f16. 4*16*16 = 1024 blocks --------
__global__ void __launch_bounds__(256)
scores_gemm_kernel(const _Float16* __restrict__ q, const _Float16* __restrict__ kk,
                   _Float16* __restrict__ sc) {
  __shared__ _Float16 lds[2 * LDS_BUF];
  int id = blockIdx.x;
  const int nt = id % (kN / 128); id /= (kN / 128);
  const int mt = id % (kN / 128); id /= (kN / 128);
  const int b  = id;
  const _Float16* A  = q  + (size_t)b * kN * kO;  // queries, row-major over O
  const _Float16* Bt = kk + (size_t)b * kN * kO;  // keys: B-col = key row
  const int row0 = mt * 128, col0 = nt * 128;
  v8f acc[8] = {};
  gemm_block_128x128<kO>(A, kO, row0, Bt, kO, col0, lds, acc);
  const int myrow0 = row0 + ((threadIdx.x >> 5) << 4);
#pragma unroll
  for (int j = 0; j < 8; ++j)
    store_c_f16(sc + (size_t)b * kN * kN, kN, myrow0, col0 + 16 * j, acc[j], 0.125f);
}

// ---- Stage 4: in-place row softmax over 2048 cols, 256 thr x 8 cols --------
__global__ void __launch_bounds__(256)
softmax_kernel(_Float16* __restrict__ sc) {
  _Float16* rp = sc + (size_t)blockIdx.x * kN;
  const int t = threadIdx.x;
  v8h vh = *(const v8h*)(rp + t * 8);
  float f[8], mx = -3.0e38f;
#pragma unroll
  for (int i = 0; i < 8; ++i) { f[i] = (float)vh[i]; mx = fmaxf(mx, f[i]); }
#pragma unroll
  for (int o = 16; o > 0; o >>= 1) mx = fmaxf(mx, __shfl_xor(mx, o, 32));
  __shared__ float red[8];
  if ((t & 31) == 0) red[t >> 5] = mx;
  __syncthreads();
  mx = red[0];
#pragma unroll
  for (int i = 1; i < 8; ++i) mx = fmaxf(mx, red[i]);
  float sum = 0.f;
#pragma unroll
  for (int i = 0; i < 8; ++i) { f[i] = __expf(f[i] - mx); sum += f[i]; }
#pragma unroll
  for (int o = 16; o > 0; o >>= 1) sum += __shfl_xor(sum, o, 32);
  __syncthreads();
  if ((t & 31) == 0) red[t >> 5] = sum;
  __syncthreads();
  sum = 0.f;
#pragma unroll
  for (int i = 0; i < 8; ++i) sum += red[i];
  const float inv = 1.f / sum;
#pragma unroll
  for (int i = 0; i < 8; ++i) vh[i] = (_Float16)(f[i] * inv);
  *(v8h*)(rp + t * 8) = vh;
}

// ---- Stage 5: out = P V  (K = 2048). 4*16*6 = 384 blocks -------------------
__global__ void __launch_bounds__(256)
pv_gemm_kernel(const _Float16* __restrict__ at, const _Float16* __restrict__ vt,
               float* __restrict__ out) {
  __shared__ _Float16 lds[2 * LDS_BUF];
  int id = blockIdx.x;
  const int nt = id % (kO / 128); id /= (kO / 128);
  const int mt = id % (kN / 128); id /= (kN / 128);
  const int b  = id;
  const _Float16* A  = at + (size_t)b * kN * kN;  // probs, row-major over keys
  const _Float16* Bt = vt + (size_t)b * kO * kN;  // V^T: B-col o reads vt[o][*]
  const int row0 = mt * 128, col0 = nt * 128;
  v8f acc[8] = {};
  gemm_block_128x128<kN>(A, kN, row0, Bt, kN, col0, lds, acc);
  const int myrow0 = row0 + ((threadIdx.x >> 5) << 4);
#pragma unroll
  for (int j = 0; j < 8; ++j)
    store_c_f32(out + (size_t)b * kN * kO, kO, myrow0, col0 + 16 * j, acc[j]);
}

// ---------------------------------------------------------------------------
extern "C" void kernel_launch(void* const* d_in, const int* in_sizes, int n_in,
                              void* d_out, int out_size, void* d_ws, size_t ws_size,
                              hipStream_t stream) {
  (void)in_sizes; (void)n_in; (void)out_size; (void)ws_size;
  const float* x = (const float*)d_in[0];
  const float* w = (const float*)d_in[1];
  float* out = (float*)d_out;

  // Workspace layout (all f16; total ~87 MB)
  char* p = (char*)d_ws;
  _Float16* xh = (_Float16*)p; p += XH_ELEMS  * sizeof(_Float16);
  _Float16* wt = (_Float16*)p; p += WT_ELEMS  * sizeof(_Float16);
  _Float16* q  = (_Float16*)p; p += QKV_ELEMS * sizeof(_Float16);
  _Float16* kk = (_Float16*)p; p += QKV_ELEMS * sizeof(_Float16);
  _Float16* vt = (_Float16*)p; p += QKV_ELEMS * sizeof(_Float16);
  _Float16* sc = (_Float16*)p; p += SC_ELEMS  * sizeof(_Float16);

  constexpr int total_prep = (int)((XH_ELEMS + WT_ELEMS) / 1024);       // 7872
  prep_kernel<<<total_prep, 256, 0, stream>>>(x, w, xh, wt);
  qkv_gemm_kernel<<<3 * kB * (kN / 128) * (kO / 128), 256, 0, stream>>>(xh, wt, q, kk, vt); // 1152
  scores_gemm_kernel<<<kB * (kN / 128) * (kN / 128), 256, 0, stream>>>(q, kk, sc);          // 1024
  softmax_kernel<<<kB * kN, 256, 0, stream>>>(sc);                                          // 8192
  pv_gemm_kernel<<<kB * (kN / 128) * (kO / 128), 256, 0, stream>>>(sc, vt, out);            // 384
}
